// Decoder_9552007266625
// MI455X (gfx1250) — matrix operand order repaired and checked by reading, not compiled, not run on hardware
//
#include <hip/hip_runtime.h>

#define DEV __device__ __forceinline__

typedef __attribute__((ext_vector_type(16))) __bf16 v16bf;
typedef __attribute__((ext_vector_type(8)))  float  v8f;

// ---------- helpers ----------
DEV unsigned short f2bf(float f) {
  unsigned u = __builtin_bit_cast(unsigned, f);
  u = (u + 0x7FFFu + ((u >> 16) & 1u)) >> 16;   // round-to-nearest-even
  return (unsigned short)u;
}
DEV float bf2f(unsigned short h) {
  unsigned u = ((unsigned)h) << 16;
  return __builtin_bit_cast(float, u);
}
DEV float tanh_hw(float x) {               // CDNA5 transcendental
  float r;
  asm("v_tanh_f32 %0, %1" : "=v"(r) : "v"(x));
  return r;
}

union ABu { v16bf v; uint4 q[2]; };

DEV v8f wmma_bf16(v16bf a, v16bf b, v8f c) {
  return __builtin_amdgcn_wmma_f32_16x16x32_bf16(false, a, false, b, (short)0, c,
                                                 false, false);
}

// Generic 16xK * Kx64 tile: one wave produces a 16(M) x 64(N) strip.
// A: row-major bf16, stride lda.  Bp: pre-packed tiles, tile (nt,kt) at
// ((nt*packKT + kt)*512) with per-lane 16 contiguous bf16.
DEV void mm4(const unsigned short* __restrict__ A, int lda, int rowBase,
             const unsigned short* __restrict__ Bp, int packKT, int ktOff,
             int numKt, int nt0, v8f acc[4]) {
  int lane = threadIdx.x & 31;
  int m = lane & 15, hi = lane >> 4;
  const unsigned short* arow = A + (size_t)(rowBase + m) * lda;
  for (int kt = 0; kt < numKt; ++kt) {
    ABu a;
    int k0 = kt * 32 + hi * 16;
    a.q[0] = *(const uint4*)(arow + k0);        // K = kt*32 + hi*16 + 0..7
    a.q[1] = *(const uint4*)(arow + k0 + 8);    // K = kt*32 + hi*16 + 8..15
#pragma unroll
    for (int i = 0; i < 4; ++i) {
      ABu b;
      const unsigned short* bp =
          Bp + ((size_t)(nt0 + i) * packKT + (ktOff + kt)) * 512 + lane * 16;
      b.q[0] = *(const uint4*)bp;
      b.q[1] = *(const uint4*)(bp + 8);
      acc[i] = wmma_bf16(a.v, b.v, acc[i]);
    }
  }
}

// ---------- one-time prep kernels ----------
// f32 -> bf16 bulk convert (4 elems / thread)
__global__ void k_cvt(const float* __restrict__ src, unsigned short* __restrict__ dst,
                      int n4) {
  int i = blockIdx.x * 256 + threadIdx.x;
  if (i >= n4) return;
  float4 f = ((const float4*)src)[i];
  unsigned lo = (unsigned)f2bf(f.x) | ((unsigned)f2bf(f.y) << 16);
  unsigned hi = (unsigned)f2bf(f.z) | ((unsigned)f2bf(f.w) << 16);
  ((uint2*)dst)[i] = make_uint2(lo, hi);
}

// Pack a K x N weight (B matrix, element B[k][n]) into WMMA tile layout.
// mode 0: Wa (K=1024,N=512), B[k][n]=w0[k*512+n]
// mode 1: Wrnn (K=1088,N=512), k<576: W_ih[n][k]=w0[n*576+k] else W_hh[n][k-576]=w1[n*512+k-576]
// mode 2: WoT (K=512,N=64), B[k][n]=Wo[n][k]=w0[n*512+k]
__global__ void k_pack(const float* __restrict__ w0, const float* __restrict__ w1,
                       unsigned short* __restrict__ dst, int K, int N, int mode) {
  int idx = blockIdx.x * 256 + threadIdx.x;
  if (idx >= K * N) return;
  int tile = idx >> 9, within = idx & 511;
  int L = within >> 4, j2 = within & 15;
  int nKT = K >> 5;
  int nt = tile / nKT, kt = tile % nKT;
  int hi = L >> 4, tt = L & 15;
  int k = kt * 32 + hi * 16 + j2;
  int n = nt * 16 + tt;
  float f;
  if (mode == 0)      f = w0[k * 512 + n];
  else if (mode == 1) f = (k < 576) ? w0[n * 576 + k] : w1[n * 512 + (k - 576)];
  else                f = w0[n * 512 + k];
  dst[idx] = f2bf(f);
}

// init x0 (sos) and h0 into A_rnn buffer 0 + h bf16 buffer
__global__ void k_init(const float* __restrict__ sos, const float* __restrict__ h0,
                       unsigned short* __restrict__ Arnn0,
                       unsigned short* __restrict__ hbf) {
  int idx = blockIdx.x * 256 + threadIdx.x;
  if (idx >= 256 * 576) return;
  int b = idx / 576, c = idx % 576;
  if (c < 64) {
    Arnn0[b * 1088 + c] = f2bf(sos[b * 64 + c]);
  } else {
    int j = c - 64;
    unsigned short v = f2bf(h0[b * 512 + j]);
    Arnn0[b * 1088 + 576 + j] = v;
    hbf[b * 512 + j] = v;
  }
}

// enc_proj = enc_bf16 @ Wa[512:1024]   (65536x512, K=512) -> bf16
__global__ void k_encproj(const unsigned short* __restrict__ encB,
                          const unsigned short* __restrict__ WaP,
                          unsigned short* __restrict__ epB) {
  int w = threadIdx.x >> 5;
  int mt = blockIdx.x;                      // 0..4095
  v8f z = {0, 0, 0, 0, 0, 0, 0, 0};
  v8f acc[4] = {z, z, z, z};
  mm4(encB, 512, mt * 16, WaP, 32, 16, 16, w * 4, acc);
  int lane = threadIdx.x & 31, t = lane & 15, hi = lane >> 4;
#pragma unroll
  for (int i = 0; i < 4; ++i) {
    int col = (w * 4 + i) * 16 + t;
#pragma unroll
    for (int j = 0; j < 8; ++j) {
      int row = mt * 16 + j + 8 * hi;
      epB[(size_t)row * 512 + col] = f2bf(acc[i][j]);
    }
  }
}

// ---------- per-step kernels ----------
// h_proj = h_t @ Wa[0:512] + ba   (256x512, f32 out)
__global__ void k_hproj(const unsigned short* __restrict__ hbf,
                        const unsigned short* __restrict__ WaP,
                        const float* __restrict__ ba, float* __restrict__ hproj) {
  int w = threadIdx.x >> 5;
  int mt = blockIdx.x;                      // 0..15
  v8f z = {0, 0, 0, 0, 0, 0, 0, 0};
  v8f acc[4] = {z, z, z, z};
  mm4(hbf, 512, mt * 16, WaP, 32, 0, 16, w * 4, acc);
  int lane = threadIdx.x & 31, t = lane & 15, hi = lane >> 4;
#pragma unroll
  for (int i = 0; i < 4; ++i) {
    int col = (w * 4 + i) * 16 + t;
#pragma unroll
    for (int j = 0; j < 8; ++j) {
      int row = mt * 16 + j + 8 * hi;
      hproj[row * 512 + col] = acc[i][j] + ba[col];
    }
  }
}

// score[s,b] = sum_h v[h] * tanh(hproj[b,h] + enc_proj[s,b,h]) ; one wave per (s,b)
__global__ void k_score(const unsigned short* __restrict__ epB,
                        const float* __restrict__ hproj,
                        const float* __restrict__ vvec,
                        float* __restrict__ score) {
  int w = threadIdx.x >> 5;
  int g = blockIdx.x * 8 + w;               // 0..65535 ; g = s*256 + b
  int b = g & 255;
  int lane = threadIdx.x & 31;
  const unsigned short* ep = epB + (size_t)g * 512 + lane * 16;
  const float* hp = hproj + b * 512 + lane * 16;
  const float* vp = vvec + lane * 16;
  uint4 e0 = *(const uint4*)ep;
  uint4 e1 = *(const uint4*)(ep + 8);
  unsigned ue[8] = {e0.x, e0.y, e0.z, e0.w, e1.x, e1.y, e1.z, e1.w};
  float acc = 0.f;
#pragma unroll
  for (int c = 0; c < 16; ++c) {
    float e = bf2f((unsigned short)(ue[c >> 1] >> ((c & 1) * 16)));
    acc = fmaf(vp[c], tanh_hw(hp[c] + e), acc);
  }
#pragma unroll
  for (int off = 16; off; off >>= 1) acc += __shfl_xor(acc, off, 32);
  if (lane == 0) score[g] = acc;
}

// softmax over s (per b): 256 blocks x 256 threads
__global__ void k_softmax(const float* __restrict__ score, float* __restrict__ attn) {
  __shared__ float red[256];
  int b = blockIdx.x, s = threadIdx.x;
  float x = score[s * 256 + b];
  red[s] = x;
  __syncthreads();
  for (int off = 128; off; off >>= 1) {
    if (s < off) red[s] = fmaxf(red[s], red[s + off]);
    __syncthreads();
  }
  float mx = red[0];
  __syncthreads();
  float e = __builtin_amdgcn_exp2f((x - mx) * 1.4426950408889634f);
  red[s] = e;
  __syncthreads();
  for (int off = 128; off; off >>= 1) {
    if (s < off) red[s] += red[s + off];
    __syncthreads();
  }
  attn[s * 256 + b] = e / red[0];
}

// context[b,h] = sum_s attn[s,b]*enc[s,b,h]; write bf16 into A_rnn cols 64..575
__global__ void k_context(const unsigned short* __restrict__ encB,
                          const float* __restrict__ attn,
                          unsigned short* __restrict__ Arnn) {
  int b = blockIdx.x >> 1;
  int h = (blockIdx.x & 1) * 256 + threadIdx.x;
  float acc = 0.f;
  for (int s = 0; s < 256; ++s) {
    float a = attn[s * 256 + b];
    float e = bf2f(encB[((size_t)(s * 256 + b)) * 512 + h]);
    acc = fmaf(a, e, acc);
  }
  Arnn[b * 1088 + 64 + h] = f2bf(acc);
}

// h_new = tanh(A_rnn @ Wrnn + b_ih + b_hh); write bf16 h + next A_rnn h-columns
__global__ void k_rnn(const unsigned short* __restrict__ Arnn,
                      const unsigned short* __restrict__ WrP,
                      const float* __restrict__ bih, const float* __restrict__ bhh,
                      unsigned short* __restrict__ hbf,
                      unsigned short* __restrict__ ArnnNext) {
  int w = threadIdx.x >> 5;
  int mt = blockIdx.x;                      // 0..15
  v8f z = {0, 0, 0, 0, 0, 0, 0, 0};
  v8f acc[4] = {z, z, z, z};
  mm4(Arnn, 1088, mt * 16, WrP, 34, 0, 34, w * 4, acc);
  int lane = threadIdx.x & 31, t = lane & 15, hi = lane >> 4;
#pragma unroll
  for (int i = 0; i < 4; ++i) {
    int col = (w * 4 + i) * 16 + t;
#pragma unroll
    for (int j = 0; j < 8; ++j) {
      int row = mt * 16 + j + 8 * hi;
      float hn = tanh_hw(acc[i][j] + bih[col] + bhh[col]);
      unsigned short hb = f2bf(hn);
      hbf[row * 512 + col] = hb;
      ArnnNext[row * 1088 + 576 + col] = hb;
    }
  }
}

// out = h_new @ Wo^T + bo  (256x64)
__global__ void k_out(const unsigned short* __restrict__ hbf,
                      const unsigned short* __restrict__ WoP,
                      const float* __restrict__ bo, float* __restrict__ outT) {
  int w = threadIdx.x >> 5;
  int mt = blockIdx.x * 8 + w;              // 0..15
  v8f z = {0, 0, 0, 0, 0, 0, 0, 0};
  v8f acc[4] = {z, z, z, z};
  mm4(hbf, 512, mt * 16, WoP, 16, 0, 16, 0, acc);
  int lane = threadIdx.x & 31, t = lane & 15, hi = lane >> 4;
#pragma unroll
  for (int i = 0; i < 4; ++i) {
    int col = i * 16 + t;
#pragma unroll
    for (int j = 0; j < 8; ++j) {
      int row = mt * 16 + j + 8 * hi;
      outT[row * 64 + col] = acc[i][j] + bo[col];
    }
  }
}

// emit preds[b,o,t] + one-hot(argmax) into next A_rnn x-columns
__global__ void k_argmax(const float* __restrict__ outT, float* __restrict__ dout,
                         unsigned short* __restrict__ ArnnNext, int t) {
  __shared__ float sv[64];
  __shared__ int si[64];
  int b = blockIdx.x, o = threadIdx.x;      // 64 threads
  float val = outT[b * 64 + o];
  dout[((size_t)(b * 64 + o)) * 63 + t] = val;
  sv[o] = val;
  si[o] = o;
  __syncthreads();
  for (int off = 32; off; off >>= 1) {
    if (o < off) {
      float v2 = sv[o + off];
      int i2 = si[o + off];
      if (v2 > sv[o] || (v2 == sv[o] && i2 < si[o])) { sv[o] = v2; si[o] = i2; }
    }
    __syncthreads();
  }
  int amax = si[0];
  ArnnNext[b * 1088 + o] = f2bf(o == amax ? 1.0f : 0.0f);
}

// ---------- host ----------
extern "C" void kernel_launch(void* const* d_in, const int* in_sizes, int n_in,
                              void* d_out, int out_size, void* d_ws, size_t ws_size,
                              hipStream_t stream) {
  const float* sos = (const float*)d_in[0];
  const float* h0  = (const float*)d_in[1];
  const float* enc = (const float*)d_in[2];
  const float* Wa  = (const float*)d_in[3];
  const float* ba  = (const float*)d_in[4];
  const float* vv  = (const float*)d_in[5];
  const float* Wih = (const float*)d_in[6];
  const float* bih = (const float*)d_in[7];
  const float* Whh = (const float*)d_in[8];
  const float* bhh = (const float*)d_in[9];
  const float* Wo  = (const float*)d_in[10];
  const float* bo  = (const float*)d_in[11];

  char* ws = (char*)d_ws;
  unsigned short* encB  = (unsigned short*)(ws + 0);           // 64 MiB
  unsigned short* epB   = (unsigned short*)(ws + 67108864);    // 64 MiB
  unsigned short* WaP   = (unsigned short*)(ws + 134217728);   // 1 MiB
  unsigned short* WrP   = (unsigned short*)(ws + 135266304);   // 1.06 MiB
  unsigned short* WoP   = (unsigned short*)(ws + 136380416);   // 64 KiB
  float*          hproj = (float*)        (ws + 136445952);    // 512 KiB
  float*          score = (float*)        (ws + 136970240);    // 256 KiB
  float*          attn  = (float*)        (ws + 137232384);    // 256 KiB
  unsigned short* Arnn0 = (unsigned short*)(ws + 137494528);   // 544 KiB
  unsigned short* Arnn1 = (unsigned short*)(ws + 138051584);   // 544 KiB
  unsigned short* hbf   = (unsigned short*)(ws + 138608640);   // 256 KiB
  float*          outT  = (float*)        (ws + 138870784);    // 64 KiB

  // one-time prep
  k_cvt<<<32768, 256, 0, stream>>>(enc, encB, 8388608);
  k_pack<<<2048, 256, 0, stream>>>(Wa, nullptr, WaP, 1024, 512, 0);
  k_pack<<<2176, 256, 0, stream>>>(Wih, Whh, WrP, 1088, 512, 1);
  k_pack<<<128, 256, 0, stream>>>(Wo, nullptr, WoP, 512, 64, 2);
  k_init<<<576, 256, 0, stream>>>(sos, h0, Arnn0, hbf);
  k_encproj<<<4096, 256, 0, stream>>>(encB, WaP, epB);

  for (int t = 0; t < 63; ++t) {
    unsigned short* Acur  = (t & 1) ? Arnn1 : Arnn0;
    unsigned short* Anext = (t & 1) ? Arnn0 : Arnn1;
    k_hproj<<<16, 256, 0, stream>>>(hbf, WaP, ba, hproj);
    k_score<<<8192, 256, 0, stream>>>(epB, hproj, vv, score);
    k_softmax<<<256, 256, 0, stream>>>(score, attn);
    k_context<<<512, 256, 0, stream>>>(encB, attn, Acur);
    k_rnn<<<16, 256, 0, stream>>>(Acur, WrP, bih, bhh, hbf, Anext);
    k_out<<<2, 256, 0, stream>>>(hbf, WoP, bo, outT);
    k_argmax<<<256, 64, 0, stream>>>(outT, (float*)d_out, Anext, t);
  }
}
